// SelectiveSSM_15169824489521
// MI455X (gfx1250) — matrix operand order repaired and checked by reading, not compiled, not run on hardware
//
#include <hip/hip_runtime.h>
#include <hip/hip_bf16.h>

// ---------------------------------------------------------------------------
// SelectiveSSM for MI455X (gfx1250), wave32, bf16 WMMA + VALU scan.
// LDS tiles are stored in WMMA *fragment layout* [tile][lane][16] so each
// operand is one contiguous 32B LDS read (2x ds_load_b128), no scalar packing.
// ---------------------------------------------------------------------------

typedef __attribute__((ext_vector_type(16))) __bf16 v16bf;
typedef __attribute__((ext_vector_type(4)))  __bf16 v4bf;
typedef __attribute__((ext_vector_type(8)))  float  v8f;

#define BATCH 1024
#define SEQ   512
#define TSD   64
#define SDIM  32
#define NSTATE 64
#define HID   256
#define NCLS  10

__device__ __forceinline__ v8f wmma_bf16(v16bf a, v16bf b, v8f c) {
    // D = A(16x32 bf16) x B(32x16 bf16) + C(16x16 f32)
    return __builtin_amdgcn_wmma_f32_16x16x32_bf16(
        /*neg_a=*/false, a, /*neg_b=*/false, b,
        /*c_mod=*/(short)0, c, /*reuse_a=*/false, /*reuse_b=*/false);
}

// Branch-free tanh: no EXEC divergence around WMMA regions.
__device__ __forceinline__ float fast_tanh(float x) {
    float ax = __builtin_fabsf(x);
    float e  = __expf(-2.0f * ax);               // e in (0, 1]
    float r  = (1.0f - e) * __builtin_amdgcn_rcpf(1.0f + e);
    return __builtin_copysignf(r, x);
}

// A fragment position for matrix element (m, k) of a 16x32 bf16 tile:
//   lane = (k>>4)*16 + m    ; elem = k & 15
// B fragment position for element (k, n) of a 32x16 bf16 tile:
//   lane = (k>>4)*16 + n    ; elem = k & 15
// C layout: lane -> col = lane&15, VGPR r -> row = r + 8*(lane>>4)

// ---------------------------------------------------------------------------
// Generic bf16-WMMA GEMM: C[M,N] = act(A[M,K] @ B[K,N] + bias[N])
// Macro-tile 32(M) x 64(N), BK=32, 256 threads = 8 waves, each wave one 16x16.
// ---------------------------------------------------------------------------
template<bool A_BF16, bool RELU, bool OUT_BF16>
__global__ __launch_bounds__(256) void gemm_kernel(
    const void* __restrict__ Aptr, const float* __restrict__ B,
    const float* __restrict__ bias, void* __restrict__ Cptr,
    int M, int N, int K)
{
    __shared__ __attribute__((aligned(32))) __bf16 ldsA[2 * 32 * 16]; // [mtile][lane][elem]
    __shared__ __attribute__((aligned(32))) __bf16 ldsB[4 * 32 * 16]; // [ntile][lane][elem]

    const int tid  = threadIdx.x;
    const int lane = tid & 31;
    const int wv   = tid >> 5;        // 0..7
    const int mt   = wv >> 2;         // 0..1
    const int nt   = wv & 3;          // 0..3
    const int m0   = blockIdx.x * 32;
    const int n0   = blockIdx.y * 64;

    v8f acc = {};
    const int nk = K >> 5;
    for (int ks = 0; ks < nk; ++ks) {
        const int k0 = ks << 5;
        // ---- stage A tile (32x32): one 4-elem group per thread ----
        {
            const int r = tid >> 3, c = (tid & 7) * 4;          // 4 consecutive k
            const int mtile = r >> 4, m = r & 15;
            const int lp = (c >> 4) * 16 + m;                   // same for all 4
            const int e0 = c & 15;                              // elems e0..e0+3
            v4bf v;
            if (A_BF16) {
                v = *(const v4bf*)((const __bf16*)Aptr + (size_t)(m0 + r) * K + k0 + c);
            } else {
                float4 f = *(const float4*)((const float*)Aptr + (size_t)(m0 + r) * K + k0 + c);
                v[0] = (__bf16)f.x; v[1] = (__bf16)f.y; v[2] = (__bf16)f.z; v[3] = (__bf16)f.w;
            }
            *(v4bf*)&ldsA[(mtile * 32 + lp) * 16 + e0] = v;
        }
        // ---- stage B tile (32x64): two float4 per thread ----
        #pragma unroll
        for (int i = 0; i < 2; ++i) {
            const int q = i * 256 + tid;
            const int r = q >> 4, c = (q & 15) * 4;             // 4 consecutive n
            float4 f = *(const float4*)(B + (size_t)(k0 + r) * N + n0 + c);
            const int ntile = c >> 4, nb = c & 15;
            const int base = ((ntile * 32 + (r >> 4) * 16 + nb) * 16) + (r & 15);
            ldsB[base + 0 * 16] = (__bf16)f.x;
            ldsB[base + 1 * 16] = (__bf16)f.y;
            ldsB[base + 2 * 16] = (__bf16)f.z;
            ldsB[base + 3 * 16] = (__bf16)f.w;
        }
        __syncthreads();
        v16bf af = *(const v16bf*)&ldsA[(mt * 32 + lane) * 16];
        v16bf bv = *(const v16bf*)&ldsB[(nt * 32 + lane) * 16];
        acc = wmma_bf16(af, bv, acc);
        __syncthreads();
    }
    // ---- epilogue ----
    const int cn = lane & 15, mh = lane >> 4;
    #pragma unroll
    for (int r = 0; r < 8; ++r) {
        int row = m0 + mt * 16 + mh * 8 + r;
        int col = n0 + nt * 16 + cn;
        float v = acc[r] + bias[col];
        if (RELU) v = v > 0.f ? v : 0.f;
        if (OUT_BF16) ((__bf16*)Cptr)[(size_t)row * N + col] = (__bf16)v;
        else          ((float*)Cptr)[(size_t)row * N + col]  = v;
    }
}

// ---------------------------------------------------------------------------
// Fused attention + Bu kernel. Block = (batch b, 32-step tile), 8 waves.
//  1) masked ts[32x64] -> LDS (fragment layout), W_a1[64x256] -> LDS
//  2) WMMA (32x64)@(64x256); branch-free tanh; dot W_a2 -> attn[32] (ds_add)
//  3) scale ts by attn in LDS; WMMA (32x64)@Bm[b]^T[64x64] -> Bu f32
// ---------------------------------------------------------------------------
__global__ __launch_bounds__(256) void attn_bu_kernel(
    const float* __restrict__ ts, const int* __restrict__ lengths,
    const float* __restrict__ Wa1, const float* __restrict__ ba1,
    const float* __restrict__ Wa2, const float* __restrict__ ba2,
    const __bf16* __restrict__ Bm, float* __restrict__ Bu)
{
    __shared__ __attribute__((aligned(32))) __bf16 ldsTs[2 * 2 * 32 * 16];  // [mtile][ks][lane][elem]
    __shared__ __attribute__((aligned(32))) __bf16 ldsW [16 * 2 * 32 * 16]; // [ntile][ks][lane][elem]
    __shared__ float attnAcc[32];

    const int b    = blockIdx.x;
    const int t0   = blockIdx.y * 32;
    const int tid  = threadIdx.x;
    const int lane = tid & 31;
    const int wv   = tid >> 5;
    const int len  = lengths[b];

    // ---- stage masked ts tile: two float4 per thread ----
    #pragma unroll
    for (int i = 0; i < 2; ++i) {
        const int q = i * 256 + tid;
        const int r = q >> 4, c = (q & 15) * 4;
        const int t = t0 + r;                       // always < SEQ
        float4 f = *(const float4*)(ts + ((size_t)b * SEQ + t) * TSD + c);
        if (t >= len) { f.x = 0.f; f.y = 0.f; f.z = 0.f; f.w = 0.f; }
        const int mtile = r >> 4, m = r & 15;
        const int ks = c >> 5, cc = c & 31;
        const int lp = (cc >> 4) * 16 + m;
        const int e0 = cc & 15;
        v4bf v; v[0] = (__bf16)f.x; v[1] = (__bf16)f.y; v[2] = (__bf16)f.z; v[3] = (__bf16)f.w;
        *(v4bf*)&ldsTs[(((mtile * 2 + ks) * 32 + lp) * 16) + e0] = v;
    }
    // ---- stage W_a1 (64x256): sixteen float4 per thread ----
    #pragma unroll
    for (int i = 0; i < 16; ++i) {
        const int q = i * 256 + tid;
        const int r = q >> 6, c = (q & 63) * 4;     // k = r, 4 consecutive n
        float4 f = *(const float4*)(Wa1 + (size_t)r * HID + c);
        const int ntile = c >> 4, nb = c & 15;
        const int ks = r >> 5, kb = (r & 31) >> 4;
        const int base = (((ntile * 2 + ks) * 32 + kb * 16 + nb) * 16) + (r & 15);
        ldsW[base + 0 * 16] = (__bf16)f.x;
        ldsW[base + 1 * 16] = (__bf16)f.y;
        ldsW[base + 2 * 16] = (__bf16)f.z;
        ldsW[base + 3 * 16] = (__bf16)f.w;
    }
    if (tid < 32) attnAcc[tid] = ba2[0] + 1e-8f;
    __syncthreads();

    // ---- GEMM1 + tanh + W_a2 reduction ----
    const int cn = lane & 15, mh = lane >> 4;
    for (int mtl = 0; mtl < 2; ++mtl) {
        for (int nti = 0; nti < 2; ++nti) {
            const int ntl = wv * 2 + nti;           // 16 n-tiles over 8 waves
            v8f acc = {};
            #pragma unroll
            for (int ks = 0; ks < 2; ++ks) {
                v16bf af = *(const v16bf*)&ldsTs[((mtl * 2 + ks) * 32 + lane) * 16];
                v16bf bv = *(const v16bf*)&ldsW [((ntl * 2 + ks) * 32 + lane) * 16];
                acc = wmma_bf16(af, bv, acc);
            }
            const int col = ntl * 16 + cn;
            const float bb = ba1[col], w2 = Wa2[col];
            #pragma unroll
            for (int r = 0; r < 8; ++r) {
                int row = mtl * 16 + mh * 8 + r;
                atomicAdd(&attnAcc[row], fast_tanh(acc[r] + bb) * w2);
            }
        }
    }
    __syncthreads();

    // ---- scale ts tile by attn (in place, fragment layout) ----
    #pragma unroll
    for (int i = 0; i < 8; ++i) {
        const int f = i * 256 + tid;                 // 2048 elems
        const int lp = (f >> 4) & 31, mtile = f >> 10;
        const int row = mtile * 16 + (lp & 15);
        ldsTs[f] = (__bf16)((float)ldsTs[f] * attnAcc[row]);
    }
    __syncthreads();

    // ---- GEMM2: (ts*attn) @ Bm[b]^T -> Bu tile ----
    {
        const int mtl = wv >> 2, ntl = wv & 3;       // 2x4 tiles, 8 waves
        v8f acc = {};
        #pragma unroll
        for (int ks = 0; ks < 2; ++ks) {
            v16bf af = *(const v16bf*)&ldsTs[((mtl * 2 + ks) * 32 + lane) * 16];
            // B[k=d, n] = Bm[b, n, d]: lane reads 16 contiguous bf16 (32B aligned)
            const __bf16* bp = Bm + (size_t)b * (NSTATE * TSD)
                                  + (size_t)(ntl * 16 + cn) * TSD + ks * 32 + (lane >> 4) * 16;
            v16bf bv = *(const v16bf*)bp;
            acc = wmma_bf16(af, bv, acc);
        }
        #pragma unroll
        for (int r = 0; r < 8; ++r) {
            int row = mtl * 16 + mh * 8 + r;
            int col = ntl * 16 + cn;
            Bu[((size_t)b * SEQ + t0 + row) * NSTATE + col] = acc[r];
        }
    }
}

// ---------------------------------------------------------------------------
// Sequential scan: s = Fm[b] @ s + Bu[b,t,:], t = 0..511.
// One block (64 threads) per batch. Fm row in registers, state in LDS.
// ---------------------------------------------------------------------------
__global__ __launch_bounds__(64) void scan_kernel(
    const float* __restrict__ Fm, const float* __restrict__ Bu,
    __bf16* __restrict__ stateOut)
{
    const int b = blockIdx.x;
    const int i = threadIdx.x;   // state row 0..63
    __shared__ __attribute__((aligned(16))) float sv[NSTATE];

    float frow[NSTATE];
    #pragma unroll
    for (int j = 0; j < NSTATE; ++j)
        frow[j] = Fm[((size_t)b * NSTATE + i) * NSTATE + j];

    sv[i] = 0.f;
    __syncthreads();

    const float* bub = Bu + (size_t)b * SEQ * NSTATE;
    for (int t = 0; t < SEQ; ++t) {
        float acc = bub[(size_t)t * NSTATE + i];
        const float4* s4 = (const float4*)sv;
        #pragma unroll
        for (int jv = 0; jv < NSTATE / 4; ++jv) {
            float4 s = s4[jv];
            acc += frow[jv * 4 + 0] * s.x;
            acc += frow[jv * 4 + 1] * s.y;
            acc += frow[jv * 4 + 2] * s.z;
            acc += frow[jv * 4 + 3] * s.w;
        }
        __syncthreads();
        sv[i] = acc;
        __syncthreads();
    }
    stateOut[(size_t)b * NSTATE + i] = (__bf16)sv[i];
}

// ---------------------------------------------------------------------------
// Final classifier output: logits = hc @ W_c2 + b_c2  (N=10, tiny VALU kernel)
// ---------------------------------------------------------------------------
__global__ __launch_bounds__(32) void logits_kernel(
    const __bf16* __restrict__ hc, const float* __restrict__ Wc2,
    const float* __restrict__ bc2, float* __restrict__ out)
{
    const int b = blockIdx.x;
    const int c = threadIdx.x;
    if (c < NCLS) {
        float acc = bc2[c];
        #pragma unroll 4
        for (int h = 0; h < HID; ++h)
            acc += (float)hc[(size_t)b * HID + h] * Wc2[(size_t)h * NCLS + c];
        out[(size_t)b * NCLS + c] = acc;
    }
}

// ---------------------------------------------------------------------------
// Launcher
// ---------------------------------------------------------------------------
extern "C" void kernel_launch(void* const* d_in, const int* in_sizes, int n_in,
                              void* d_out, int out_size, void* d_ws, size_t ws_size,
                              hipStream_t stream) {
    (void)in_sizes; (void)n_in; (void)out_size; (void)ws_size;

    const float* ts_values = (const float*)d_in[0];
    // d_in[1] ts_indicators, d_in[2] ts_times: unused by reference
    const float* static_f  = (const float*)d_in[3];
    const int*   lengths   = (const int*)d_in[4];
    const float* W_F1 = (const float*)d_in[5];  const float* b_F1 = (const float*)d_in[6];
    const float* W_F2 = (const float*)d_in[7];  const float* b_F2 = (const float*)d_in[8];
    const float* W_B1 = (const float*)d_in[9];  const float* b_B1 = (const float*)d_in[10];
    const float* W_B2 = (const float*)d_in[11]; const float* b_B2 = (const float*)d_in[12];
    const float* W_a1 = (const float*)d_in[13]; const float* b_a1 = (const float*)d_in[14];
    const float* W_a2 = (const float*)d_in[15]; const float* b_a2 = (const float*)d_in[16];
    const float* W_c1 = (const float*)d_in[17]; const float* b_c1 = (const float*)d_in[18];
    const float* W_c2 = (const float*)d_in[19]; const float* b_c2 = (const float*)d_in[20];
    float* logits = (float*)d_out;

    // workspace layout (all offsets 256B-aligned)
    char* w = (char*)d_ws;
    __bf16* hF = (__bf16*)(w + 0);                       //  512 KB
    __bf16* hB = (__bf16*)(w + (size_t)512 * 1024);      //  512 KB
    float*  Fm = (float*) (w + (size_t)1  * 1024 * 1024);            // 16 MB
    __bf16* Bm = (__bf16*)(w + (size_t)17 * 1024 * 1024);            //  8 MB
    float*  Bu = (float*) (w + (size_t)25 * 1024 * 1024);            // 128 MB
    __bf16* st = (__bf16*)(w + (size_t)153 * 1024 * 1024);           // 128 KB
    __bf16* hc = (__bf16*)(w + (size_t)153 * 1024 * 1024 + 256 * 1024); // 512 KB

    dim3 blk(256);

    // h_F / h_B : [1024,32] @ [32,256] + bias, relu -> bf16
    gemm_kernel<false, true, true><<<dim3(BATCH / 32, HID / 64), blk, 0, stream>>>(
        static_f, W_F1, b_F1, hF, BATCH, HID, SDIM);
    gemm_kernel<false, true, true><<<dim3(BATCH / 32, HID / 64), blk, 0, stream>>>(
        static_f, W_B1, b_B1, hB, BATCH, HID, SDIM);

    // Fm (f32 out, for the scan) / Bm (bf16 out, consumed by WMMA in attn_bu)
    gemm_kernel<true, false, false><<<dim3(BATCH / 32, 4096 / 64), blk, 0, stream>>>(
        hF, W_F2, b_F2, Fm, BATCH, NSTATE * NSTATE, HID);
    gemm_kernel<true, false, true><<<dim3(BATCH / 32, 4096 / 64), blk, 0, stream>>>(
        hB, W_B2, b_B2, Bm, BATCH, NSTATE * TSD, HID);

    // fused attention + Bu
    attn_bu_kernel<<<dim3(BATCH, SEQ / 32), blk, 0, stream>>>(
        ts_values, lengths, W_a1, b_a1, W_a2, b_a2, Bm, Bu);

    // sequential scan over T=512
    scan_kernel<<<dim3(BATCH), dim3(64), 0, stream>>>(Fm, Bu, st);

    // classifier hidden: [1024,64] @ [64,256] + bias, relu -> bf16
    gemm_kernel<true, true, true><<<dim3(BATCH / 32, HID / 64), blk, 0, stream>>>(
        st, W_c1, b_c1, hc, BATCH, HID, NSTATE);

    // logits
    logits_kernel<<<dim3(BATCH), dim3(32), 0, stream>>>(hc, W_c2, b_c2, logits);
}
